// RelativeAttention_72911364817444
// MI455X (gfx1250) — matrix operand order repaired and checked by reading, not compiled, not run on hardware
//
#include <hip/hip_runtime.h>
#include <hip/hip_bf16.h>

typedef __attribute__((ext_vector_type(16))) _Float16 v16h;
typedef __attribute__((ext_vector_type(8)))  _Float16 v8h;
typedef __attribute__((ext_vector_type(8)))  float    v8f;

union V16 { v16h v; v8h h[2]; };

static __device__ __forceinline__ v8f zero8f() {
  v8f c;
#pragma unroll
  for (int i = 0; i < 8; ++i) c[i] = 0.0f;
  return c;
}

// ---------------- sizes ----------------
#define NB     512
#define NHEADS 8
#define DK     32
#define OCH    768         // NHEADS*(2*DK+DV)
#define CIN    256
// Ft is halo-padded NHWC: [B][10][10][C] f16, zero border
#define WT_BYTES  (9u * OCH * CIN * 2u)               // 3,538,944
#define FT_BYTES  ((unsigned)NB * 100u * CIN * 2u)    // 26,214,400

// -------- pre-pass: weights [768,256,3,3] f32 -> Wt[kk][o][c] f16 --------
__global__ __launch_bounds__(256) void relattn_wt_kernel(
    const float* __restrict__ w, _Float16* __restrict__ Wt) {
  int idx = blockIdx.x * 256 + threadIdx.x;
  if (idx >= OCH * CIN * 9) return;
  int o  = idx / (CIN * 9);
  int r  = idx - o * (CIN * 9);
  int c  = r / 9;
  int kk = r - c * 9;
  Wt[((kk * OCH + o) << 8) + c] = (_Float16)w[idx];
}

// -------- pre-pass: feature NCHW f32 -> halo-padded NHWC f16 --------
__global__ __launch_bounds__(256) void relattn_ft_kernel(
    const float* __restrict__ f, _Float16* __restrict__ Ft) {
  int idx = blockIdx.x * 256 + threadIdx.x;   // exactly 512*100*256 threads
  int c = idx & 255;
  int p = idx >> 8;                            // b*100 + yy*10 + xx
  int b = p / 100;
  int r = p - b * 100;
  int yy = r / 10;
  int xx = r - yy * 10;
  float v = 0.0f;
  if (yy >= 1 && yy <= 8 && xx >= 1 && xx <= 8)
    v = f[(((b << 8) + c) << 6) + ((yy - 1) << 3) + (xx - 1)];
  Ft[idx] = (_Float16)v;
}

// -------- conv 3x3 SAME as implicit GEMM via WMMA f16 --------
// M = 768 (out ch), N = 32768 (b*64+y*8+x), K = 2304 (tap-major, c inner)
// grid: (512 Ntiles64, 6 Mtiles128), block 256 (8 waves), wave tile 16M x 64N
// Fully unrolled, branch-free: all tap/chunk offsets fold into load immediates.
__global__ __launch_bounds__(256) void relattn_conv_kernel(
    const _Float16* __restrict__ Wt, const _Float16* __restrict__ Ft,
    _Float16* __restrict__ qkv16) {
  const int lane  = threadIdx.x & 31;
  const int wave  = threadIdx.x >> 5;
  const int lr    = lane & 15;
  const int hi    = lane >> 4;                 // 0/1 half-wave
  const int mBase = blockIdx.y * 128 + wave * 16;
  const int nBase = blockIdx.x * 64;

  // A rows: lane<16 holds K{0..7,16..23}; lane>=16 shifted by +8
  const _Float16* wbase = Wt + ((mBase + lr) << 8) + hi * 8;

  // B cols: same K order as A: lanes<16 K{0..7,16..23}; lanes>=16 shifted by +8
  const _Float16* fptr[4];
#pragma unroll
  for (int nt = 0; nt < 4; ++nt) {
    int n = nBase + nt * 16 + lr;
    int b = n >> 6, y = (n >> 3) & 7, x = n & 7;
    // halo pad: interior pixel (y,x) sits at padded (y+1,x+1); tap offset adds (kh*10+kw)
    fptr[nt] = Ft + ((b * 100 + y * 10 + x) << 8) + hi * 8;
  }

  v8f acc[4];
#pragma unroll
  for (int nt = 0; nt < 4; ++nt) acc[nt] = zero8f();

#pragma unroll
  for (int kk = 0; kk < 9; ++kk) {
    const int kh = kk / 3, kw = kk - kh * 3;
    const int woff = kk * (OCH << 8);           // compile-time immediates
    const int foff = (kh * 10 + kw) << 8;
#pragma unroll
    for (int cb = 0; cb < 256; cb += 32) {
      V16 A;
      A.h[0] = *(const v8h*)(wbase + woff + cb);
      A.h[1] = *(const v8h*)(wbase + woff + cb + 16);
#pragma unroll
      for (int nt = 0; nt < 4; ++nt) {
        V16 Bf;
        Bf.h[0] = *(const v8h*)(fptr[nt] + foff + cb);
        Bf.h[1] = *(const v8h*)(fptr[nt] + foff + cb + 16);
        acc[nt] = __builtin_amdgcn_wmma_f32_16x16x32_f16(
            false, A.v, false, Bf.v, (short)0, acc[nt], false, false);
      }
    }
  }

  // C/D layout: lane col = lr, rows = hi*8 + i
#pragma unroll
  for (int nt = 0; nt < 4; ++nt) {
#pragma unroll
    for (int i = 0; i < 8; ++i) {
      int o = mBase + hi * 8 + i;
      int n = nBase + nt * 16 + lr;
      int b = n >> 6, l = n & 63;
      qkv16[((b * OCH + o) << 6) + l] = (_Float16)acc[nt][i];
    }
  }
}

// -------- attention: one block per (b, head) --------
#define QT_S  40   // 32 + 8 pad (halfs), 80B row -> 16B aligned
#define V_S   72   // 64 + 8 pad (halfs)
#define S_S   65   // f32, conflict-free column sweeps
#define P_S   72   // f16 softmaxed probs

__global__ __launch_bounds__(256) void relattn_attn_kernel(
    const _Float16* __restrict__ qkv16,
    const float* __restrict__ wEmb, const float* __restrict__ hEmb,
    float* __restrict__ out) {
  __shared__ __attribute__((aligned(16))) _Float16 Qt[64 * QT_S];
  __shared__ __attribute__((aligned(16))) _Float16 Kt[64 * QT_S];
  __shared__ __attribute__((aligned(16))) _Float16 Vs[32 * V_S];
  __shared__ __attribute__((aligned(16))) float    Sm[64 * S_S];
  __shared__ __attribute__((aligned(16))) _Float16 P16[64 * P_S];
  __shared__ __attribute__((aligned(16))) float Pw[64 * 16];
  __shared__ __attribute__((aligned(16))) float Ph[64 * 16];
  __shared__ __attribute__((aligned(16))) float Ew[15 * 32];
  __shared__ __attribute__((aligned(16))) float Eh[15 * 32];

  const int tid  = threadIdx.x;
  const int bh   = blockIdx.x;
  const int b    = bh >> 3, h = bh & 7;
  const int qoff = b * OCH + h * DK;
  const int koff = qoff + 256;
  const int voff = qoff + 512;

  for (int t = tid; t < 2048; t += 256) {
    int d = t >> 6, l = t & 63;
    Qt[l * QT_S + d] = qkv16[((qoff + d) << 6) + l];
    Kt[l * QT_S + d] = qkv16[((koff + d) << 6) + l];
    Vs[d * V_S + l]  = qkv16[((voff + d) << 6) + l];
  }
  for (int t = tid; t < 480; t += 256) { Ew[t] = wEmb[t]; Eh[t] = hEmb[t]; }
  __syncthreads();

  // rel tables: Pw[j][m] = sum_d Qt[j][d]*wEmb[m][d], same for Ph/hEmb
  for (int t = tid; t < 2048; t += 256) {
    int which = t >> 10;
    int j = (t >> 4) & 63;
    int m = t & 15;
    if (m < 15) {
      const float* E = which ? Eh : Ew;
      float s = 0.f;
#pragma unroll
      for (int d = 0; d < 32; ++d) s += (float)Qt[j * QT_S + d] * E[m * 32 + d];
      (which ? Ph : Pw)[j * 16 + m] = s;
    }
  }
  __syncthreads();

  const int lane = tid & 31, wave = tid >> 5;
  const int lr = lane & 15, hi = lane >> 4;

  // S = Q^T K  (16 tiles, 2 per wave), + rel skew, * 1/sqrt(32)
#pragma unroll
  for (int t0 = 0; t0 < 2; ++t0) {
    int t = wave + t0 * 8;
    int mt = t >> 2, nt = t & 3;
    V16 A, Bf;
    const _Float16* ap = &Qt[(mt * 16 + lr) * QT_S + hi * 8];
    A.h[0] = *(const v8h*)(ap);
    A.h[1] = *(const v8h*)(ap + 16);
    const _Float16* bp = &Kt[(nt * 16 + lr) * QT_S + hi * 8];
    Bf.h[0] = *(const v8h*)(bp);
    Bf.h[1] = *(const v8h*)(bp + 16);
    v8f c = zero8f();
    c = __builtin_amdgcn_wmma_f32_16x16x32_f16(
        false, A.v, false, Bf.v, (short)0, c, false, false);
#pragma unroll
    for (int i = 0; i < 8; ++i) {
      int l = mt * 16 + hi * 8 + i;
      int m = nt * 16 + lr;
      int yq = l >> 3, xq = l & 7, yk = m >> 3, xk = m & 7;
      float s = c[i]
              + Pw[((yq << 3) + xk) * 16 + (7 + yk - xk)]
              + Ph[((xq << 3) + yq) * 16 + (7 + xk - xq)];
      Sm[l * S_S + m] = s * 0.17677669529663687f;  // 1/sqrt(32)
    }
  }
  __syncthreads();

  // softmax rows (threads 0..63)
  if (tid < 64) {
    const float* row = &Sm[tid * S_S];
    float mx = -1e30f;
    for (int m = 0; m < 64; ++m) mx = fmaxf(mx, row[m]);
    float sum = 0.f;
    for (int m = 0; m < 64; ++m) sum += __expf(row[m] - mx);
    float inv = 1.0f / sum;
    for (int m = 0; m < 64; ++m)
      P16[tid * P_S + m] = (_Float16)(__expf(row[m] - mx) * inv);
  }
  __syncthreads();

  // O = P * V   (M=64, N=32, K=64): 8 tiles, one per wave, 2 K-steps
  {
    int mt = wave >> 1, nt = wave & 1;
    v8f c = zero8f();
#pragma unroll
    for (int ks = 0; ks < 2; ++ks) {
      V16 A, Bf;
      const _Float16* ap = &P16[(mt * 16 + lr) * P_S + ks * 32 + hi * 8];
      A.h[0] = *(const v8h*)(ap);
      A.h[1] = *(const v8h*)(ap + 16);
      const _Float16* bp = &Vs[(nt * 16 + lr) * V_S + ks * 32 + hi * 8];
      Bf.h[0] = *(const v8h*)(bp);
      Bf.h[1] = *(const v8h*)(bp + 16);
      c = __builtin_amdgcn_wmma_f32_16x16x32_f16(
          false, A.v, false, Bf.v, (short)0, c, false, false);
    }
#pragma unroll
    for (int i = 0; i < 8; ++i) {
      int l = mt * 16 + hi * 8 + i;
      int d = nt * 16 + lr;
      out[(((bh) << 6) + l) * 32 + d] = c[i];
    }
  }
}

extern "C" void kernel_launch(void* const* d_in, const int* in_sizes, int n_in,
                              void* d_out, int out_size, void* d_ws, size_t ws_size,
                              hipStream_t stream) {
  (void)in_sizes; (void)n_in; (void)out_size; (void)ws_size;
  const float* feature = (const float*)d_in[0];
  const float* wqkv    = (const float*)d_in[1];
  const float* wEmb    = (const float*)d_in[2];
  const float* hEmb    = (const float*)d_in[3];
  float* out = (float*)d_out;

  char* ws = (char*)d_ws;
  _Float16* Wt    = (_Float16*)(ws);
  _Float16* Ft    = (_Float16*)(ws + WT_BYTES);
  _Float16* qkv16 = (_Float16*)(ws + WT_BYTES + FT_BYTES);

  relattn_wt_kernel<<<(OCH * CIN * 9 + 255) / 256, 256, 0, stream>>>(wqkv, Wt);
  relattn_ft_kernel<<<(NB * 100 * CIN) / 256, 256, 0, stream>>>(feature, Ft);
  relattn_conv_kernel<<<dim3(512, 6, 1), 256, 0, stream>>>(Wt, Ft, qkv16);
  relattn_attn_kernel<<<NB * NHEADS, 256, 0, stream>>>(qkv16, wEmb, hEmb, out);
}
